// ResMamba_63720134803786
// MI455X (gfx1250) — matrix operand order repaired and checked by reading, not compiled, not run on hardware
//
#include <hip/hip_runtime.h>
#include <hip/hip_bf16.h>
#include <math.h>
#include <stdint.h>

#define D_MODEL  256
#define D_STATE  64
#define D_CONV   4
#define HEADDIM  64
#define D_INNER  512
#define NHEADS   8
#define D_XBC    640            // D_INNER + 2*D_STATE
#define D_IN_PROJ 1160          // 2*D_INNER + 2*D_STATE + NHEADS
#define NPAD1    1216           // D_IN_PROJ padded to 19*64 (4 tiles/wave, no guards)
#define BATCH    64
#define SEQ      250
#define ROWS     (BATCH*SEQ)    // 16000

typedef __attribute__((ext_vector_type(16))) __bf16 v16bf;
typedef __attribute__((ext_vector_type(8)))  __bf16 v8bf;
typedef __attribute__((ext_vector_type(8)))  float  v8f;

static __device__ __forceinline__ __bf16 f2bf(float f) {
    unsigned u = __builtin_bit_cast(unsigned, f);
    u += 0x7FFFu + ((u >> 16) & 1u);           // round-to-nearest-even
    unsigned short s = (unsigned short)(u >> 16);
    return __builtin_bit_cast(__bf16, s);
}

static __device__ __forceinline__ v16bf cat8(v8bf lo, v8bf hi) {
    return __builtin_shufflevector(lo, hi, 0,1,2,3,4,5,6,7,8,9,10,11,12,13,14,15);
}

// Async global -> LDS copy (CDNA5, tracked by ASYNCcnt). lds_off is the
// 32-bit LDS byte offset (generic shared-pointer truncated to its low 32 bits).
static __device__ __forceinline__ void async_ld_b128(unsigned lds_off, const void* gptr) {
    asm volatile("global_load_async_to_lds_b128 %0, %1, off"
                 :: "v"(lds_off), "v"(gptr) : "memory");
}
static __device__ __forceinline__ void wait_async0() {
    asm volatile("s_wait_asynccnt 0x0" ::: "memory");
}

// ---------------------------------------------------------------------------
// One-time operand prep (off the critical path; everything stays in L2).
// ---------------------------------------------------------------------------
__global__ void k_cvt_u(const float* __restrict__ x, __bf16* __restrict__ ubf) {
    int idx = blockIdx.x * blockDim.x + threadIdx.x;     // ROWS*256
    if (idx >= ROWS * 256) return;
    int k = idx & 255, r = idx >> 8;
    ubf[idx] = (k < 240) ? f2bf(x[(size_t)r * 240 + k]) : __builtin_bit_cast(__bf16, (unsigned short)0);
}
__global__ void k_cvt_win(const float* __restrict__ Win, __bf16* __restrict__ WinT) {
    int idx = blockIdx.x * blockDim.x + threadIdx.x;     // NPAD1*256
    if (idx >= NPAD1 * 256) return;
    int k = idx & 255, n = idx >> 8;
    WinT[idx] = (n < D_IN_PROJ) ? f2bf(Win[(size_t)k * D_IN_PROJ + n])
                                : __builtin_bit_cast(__bf16, (unsigned short)0);
}
__global__ void k_cvt_wout(const float* __restrict__ Wout, __bf16* __restrict__ WoutT) {
    int idx = blockIdx.x * blockDim.x + threadIdx.x;     // 256*512
    if (idx >= 256 * 512) return;
    int k = idx & 511, n = idx >> 9;
    WoutT[idx] = f2bf(Wout[(size_t)k * D_MODEL + n]);
}

// ---------------------------------------------------------------------------
// GEMM1: zxbcdt[16000 x 1160] = u @ W_in. 4 waves/block; each wave: one M-tile
// x four N-tiles. Shared B panel (64 cols x 256 K bf16 = 32 KB, contiguous in
// WinT) is staged once per block into LDS via async copies; A streams from L2.
// grid = (19, 250)
// ---------------------------------------------------------------------------
__global__ __launch_bounds__(128) void k_inproj(const __bf16* __restrict__ ubf,
                                                const __bf16* __restrict__ WinT,
                                                float* __restrict__ zxbcdt) {
    __shared__ __bf16 sBp[64 * 256];               // 32 KB B panel
    const int lane  = threadIdx.x & 31;
    const int wave  = threadIdx.x >> 5;
    const int mTile = blockIdx.y * 4 + wave;       // 0..999
    const int nBase = blockIdx.x * 64;             // 4 tiles of 16
    const int r16   = lane & 15;
    const int khalf = lane >> 4;
    const int row   = mTile * 16 + r16;

    // ---- stage B panel: 2048 x 16B chunks, 16 per thread, contiguous ----
    {
        const char* gB = (const char*)(WinT + (size_t)nBase * 256);
        unsigned ldsBase = (unsigned)(uintptr_t)&sBp[0];
        #pragma unroll
        for (int i = 0; i < 16; ++i) {
            int chunk = threadIdx.x + i * 128;
            async_ld_b128(ldsBase + chunk * 16, gB + (size_t)chunk * 16);
        }
        wait_async0();
    }
    __syncthreads();

    const __bf16* arow = ubf + (size_t)row * 256;
    v8f acc0 = {}, acc1 = {}, acc2 = {}, acc3 = {};

    for (int k0 = 0; k0 < D_MODEL; k0 += 32) {
        const int kaL = k0 + khalf * 16;           // A: lanes0-15 K0..15; lanes16-31 K16..31
        v16bf a = cat8(*(const v8bf*)(arow + kaL), *(const v8bf*)(arow + kaL + 8));
        __builtin_prefetch((const void*)(arow + kaL + 32), 0, 3);
        const int kb = k0 + khalf * 16;            // B: lanes0-15 K0..15; lanes16-31 K16..31
        const __bf16* b0p = sBp + (      r16) * 256 + kb;
        const __bf16* b1p = sBp + (16 + r16) * 256 + kb;
        const __bf16* b2p = sBp + (32 + r16) * 256 + kb;
        const __bf16* b3p = sBp + (48 + r16) * 256 + kb;
        v16bf bm0 = cat8(*(const v8bf*)b0p, *(const v8bf*)(b0p + 8));
        v16bf bm1 = cat8(*(const v8bf*)b1p, *(const v8bf*)(b1p + 8));
        v16bf bm2 = cat8(*(const v8bf*)b2p, *(const v8bf*)(b2p + 8));
        v16bf bm3 = cat8(*(const v8bf*)b3p, *(const v8bf*)(b3p + 8));
        acc0 = __builtin_amdgcn_wmma_f32_16x16x32_bf16(false, a, false, bm0, (short)0, acc0, false, false);
        acc1 = __builtin_amdgcn_wmma_f32_16x16x32_bf16(false, a, false, bm1, (short)0, acc1, false, false);
        acc2 = __builtin_amdgcn_wmma_f32_16x16x32_bf16(false, a, false, bm2, (short)0, acc2, false, false);
        acc3 = __builtin_amdgcn_wmma_f32_16x16x32_bf16(false, a, false, bm3, (short)0, acc3, false, false);
    }
    #pragma unroll
    for (int j = 0; j < 8; ++j) {
        const int m = mTile * 16 + j + khalf * 8;
        float* orow = zxbcdt + (size_t)m * D_IN_PROJ;
        int c0 = nBase + r16;
        if (c0      < D_IN_PROJ) orow[c0]      = acc0[j];
        if (c0 + 16 < D_IN_PROJ) orow[c0 + 16] = acc1[j];
        if (c0 + 32 < D_IN_PROJ) orow[c0 + 32] = acc2[j];
        if (c0 + 48 < D_IN_PROJ) orow[c0 + 48] = acc3[j];
    }
}

// ---------------------------------------------------------------------------
// Depthwise causal conv(4) + bias + SiLU (branch-free)
// ---------------------------------------------------------------------------
__global__ void k_conv(const float* __restrict__ zxbcdt,
                       const float* __restrict__ conv_w,
                       const float* __restrict__ conv_b,
                       float* __restrict__ xbc) {
    int idx = blockIdx.x * blockDim.x + threadIdx.x;
    if (idx >= ROWS * D_XBC) return;
    int c = idx % D_XBC;
    int r = idx / D_XBC;
    int t = r % SEQ;
    int rbase = r - t;
    float acc = conv_b[c];
    #pragma unroll
    for (int k = 0; k < D_CONV; ++k) {
        int tt  = t + k - (D_CONV - 1);
        int tcl = (tt < 0) ? 0 : tt;
        float msk = (tt < 0) ? 0.f : 1.f;
        acc += msk * zxbcdt[(size_t)(rbase + tcl) * D_IN_PROJ + D_INNER + c] * conv_w[c * D_CONV + k];
    }
    xbc[(size_t)r * D_XBC + c] = acc / (1.f + __expf(-acc));   // SiLU
}

// ---------------------------------------------------------------------------
// dt = softplus(dt_raw + dt_bias)
// ---------------------------------------------------------------------------
__global__ void k_dt(const float* __restrict__ zxbcdt,
                     const float* __restrict__ dt_bias,
                     float* __restrict__ dts) {
    int idx = blockIdx.x * blockDim.x + threadIdx.x;
    if (idx >= ROWS * NHEADS) return;
    int h = idx % NHEADS;
    int r = idx / NHEADS;
    float v = zxbcdt[(size_t)r * D_IN_PROJ + D_INNER + D_XBC + h] + dt_bias[h];
    dts[idx] = (v > 20.f) ? v : log1pf(__expf(v));
}

// ---------------------------------------------------------------------------
// Selective scan: 512 workgroups = (batch, head). 64x64 fp32 state in regs.
// ---------------------------------------------------------------------------
__global__ __launch_bounds__(256) void k_scan(const float* __restrict__ xbc,
                                              const float* __restrict__ dts,
                                              const float* __restrict__ A_log,
                                              const float* __restrict__ Dp,
                                              float* __restrict__ yscan) {
    const int bh = blockIdx.x;
    const int b  = bh >> 3;
    const int h  = bh & 7;
    const int tid = threadIdx.x;
    const int p  = tid & 63;
    const int n0 = (tid >> 6) * 16;
    const float Ah = -__expf(A_log[h]);
    const float Dh = Dp[h];

    __shared__ float sB[D_STATE], sC[D_STATE], sX[HEADDIM], sPart[256];
    __shared__ float sDt;

    float hreg[16];
    #pragma unroll
    for (int i = 0; i < 16; ++i) hreg[i] = 0.f;

    for (int t = 0; t < SEQ; ++t) {
        const int r = b * SEQ + t;
        if (tid < 64)        sB[tid]       = xbc[(size_t)r * D_XBC + D_INNER + tid];
        else if (tid < 128)  sC[tid - 64]  = xbc[(size_t)r * D_XBC + D_INNER + 64 + (tid - 64)];
        else if (tid < 192)  sX[tid - 128] = xbc[(size_t)r * D_XBC + h * HEADDIM + (tid - 128)];
        else if (tid == 192) sDt           = dts[r * NHEADS + h];
        __syncthreads();

        const float dtv   = sDt;
        const float decay = __expf(dtv * Ah);
        const float coef  = dtv * sX[p];
        float part = 0.f;
        #pragma unroll
        for (int i = 0; i < 16; ++i) {
            hreg[i] = hreg[i] * decay + coef * sB[n0 + i];
            part   += hreg[i] * sC[n0 + i];
        }
        sPart[tid] = part;
        __syncthreads();
        if (tid < 64) {
            float y = sPart[tid] + sPart[tid + 64] + sPart[tid + 128] + sPart[tid + 192]
                      + Dh * sX[tid];
            yscan[(size_t)r * D_INNER + h * HEADDIM + tid] = y;
        }
        __syncthreads();
    }
}

// ---------------------------------------------------------------------------
// Gate with silu(z), RMS-norm, scale by norm_w; emit bf16 for GEMM2.
// ---------------------------------------------------------------------------
__global__ __launch_bounds__(256) void k_gatenorm(const float* __restrict__ yscan,
                                                  const float* __restrict__ zxbcdt,
                                                  const float* __restrict__ norm_w,
                                                  __bf16* __restrict__ ynorm) {
    const int r = blockIdx.x;
    const int tid = threadIdx.x;
    __shared__ float red[256];
    float v[2];
    float ss = 0.f;
    #pragma unroll
    for (int j = 0; j < 2; ++j) {
        int c = tid + j * 256;
        float z  = zxbcdt[(size_t)r * D_IN_PROJ + c];
        float sz = z / (1.f + __expf(-z));
        float y  = yscan[(size_t)r * D_INNER + c] * sz;
        v[j] = y;
        ss += y * y;
    }
    red[tid] = ss;
    __syncthreads();
    for (int s = 128; s > 0; s >>= 1) {
        if (tid < s) red[tid] += red[tid + s];
        __syncthreads();
    }
    const float scale = rsqrtf(red[0] / (float)D_INNER + 1e-5f);
    #pragma unroll
    for (int j = 0; j < 2; ++j) {
        int c = tid + j * 256;
        ynorm[(size_t)r * D_INNER + c] = f2bf(v[j] * scale * norm_w[c]);
    }
}

// ---------------------------------------------------------------------------
// GEMM2: out = selu(ynorm @ W_out)[:, :240]; B panel (64 x 512 bf16 = 64 KB)
// staged in LDS via async copies. grid (4, 250).
// ---------------------------------------------------------------------------
__global__ __launch_bounds__(128) void k_outproj(const __bf16* __restrict__ ynorm,
                                                 const __bf16* __restrict__ WoutT,
                                                 float* __restrict__ out) {
    __shared__ __bf16 sBp[64 * 512];               // 64 KB B panel
    const int lane  = threadIdx.x & 31;
    const int wave  = threadIdx.x >> 5;
    const int mTile = blockIdx.y * 4 + wave;
    const int nBase = blockIdx.x * 64;             // 0,64,128,192
    const int r16   = lane & 15;
    const int khalf = lane >> 4;
    const int row   = mTile * 16 + r16;

    {
        const char* gB = (const char*)(WoutT + (size_t)nBase * 512);
        unsigned ldsBase = (unsigned)(uintptr_t)&sBp[0];
        #pragma unroll
        for (int i = 0; i < 32; ++i) {
            int chunk = threadIdx.x + i * 128;     // 4096 x 16B chunks
            async_ld_b128(ldsBase + chunk * 16, gB + (size_t)chunk * 16);
        }
        wait_async0();
    }
    __syncthreads();

    const __bf16* arow = ynorm + (size_t)row * D_INNER;
    v8f acc0 = {}, acc1 = {}, acc2 = {}, acc3 = {};

    for (int k0 = 0; k0 < D_INNER; k0 += 32) {
        const int kaL = k0 + khalf * 16;
        v16bf a = cat8(*(const v8bf*)(arow + kaL), *(const v8bf*)(arow + kaL + 8));
        __builtin_prefetch((const void*)(arow + kaL + 32), 0, 3);
        const int kb = k0 + khalf * 16;
        const __bf16* b0p = sBp + (      r16) * 512 + kb;
        const __bf16* b1p = sBp + (16 + r16) * 512 + kb;
        const __bf16* b2p = sBp + (32 + r16) * 512 + kb;
        const __bf16* b3p = sBp + (48 + r16) * 512 + kb;
        v16bf bm0 = cat8(*(const v8bf*)b0p, *(const v8bf*)(b0p + 8));
        v16bf bm1 = cat8(*(const v8bf*)b1p, *(const v8bf*)(b1p + 8));
        v16bf bm2 = cat8(*(const v8bf*)b2p, *(const v8bf*)(b2p + 8));
        v16bf bm3 = cat8(*(const v8bf*)b3p, *(const v8bf*)(b3p + 8));
        acc0 = __builtin_amdgcn_wmma_f32_16x16x32_bf16(false, a, false, bm0, (short)0, acc0, false, false);
        acc1 = __builtin_amdgcn_wmma_f32_16x16x32_bf16(false, a, false, bm1, (short)0, acc1, false, false);
        acc2 = __builtin_amdgcn_wmma_f32_16x16x32_bf16(false, a, false, bm2, (short)0, acc2, false, false);
        acc3 = __builtin_amdgcn_wmma_f32_16x16x32_bf16(false, a, false, bm3, (short)0, acc3, false, false);
    }
    const float SELU_L = 1.0507009873554805f;
    const float SELU_A = 1.6732632423543772f;
    #pragma unroll
    for (int j = 0; j < 8; ++j) {
        const int m = mTile * 16 + j + khalf * 8;
        float* orow = out + (size_t)m * 240;
        const int c0 = nBase + r16;
        float vv[4] = { acc0[j], acc1[j], acc2[j], acc3[j] };
        #pragma unroll
        for (int nt = 0; nt < 4; ++nt) {
            int c = c0 + nt * 16;
            if (c < 240) {
                float v = vv[nt];
                orow[c] = (v > 0.f) ? SELU_L * v : SELU_L * SELU_A * (__expf(v) - 1.f);
            }
        }
    }
}

// ---------------------------------------------------------------------------
extern "C" void kernel_launch(void* const* d_in, const int* in_sizes, int n_in,
                              void* d_out, int out_size, void* d_ws, size_t ws_size,
                              hipStream_t stream) {
    (void)in_sizes; (void)n_in; (void)out_size; (void)ws_size;
    const float* x      = (const float*)d_in[0];
    const float* W_in   = (const float*)d_in[1];
    const float* conv_w = (const float*)d_in[2];
    const float* conv_b = (const float*)d_in[3];
    const float* A_log  = (const float*)d_in[4];
    const float* dt_b   = (const float*)d_in[5];
    const float* Dp     = (const float*)d_in[6];
    const float* nw     = (const float*)d_in[7];
    const float* W_out  = (const float*)d_in[8];
    float* out = (float*)d_out;

    char* ws = (char*)d_ws;
    size_t off = 0;
    float*  zxbcdt = (float*)(ws + off);  off += (size_t)ROWS * D_IN_PROJ * 4;
    float*  xbc    = (float*)(ws + off);  off += (size_t)ROWS * D_XBC * 4;
    float*  dts    = (float*)(ws + off);  off += (size_t)ROWS * NHEADS * 4;
    float*  yscan  = (float*)(ws + off);  off += (size_t)ROWS * D_INNER * 4;
    __bf16* ynorm  = (__bf16*)(ws + off); off += (size_t)ROWS * D_INNER * 2;
    __bf16* ubf    = (__bf16*)(ws + off); off += (size_t)ROWS * 256 * 2;
    __bf16* WinT   = (__bf16*)(ws + off); off += (size_t)NPAD1 * 256 * 2;
    __bf16* WoutT  = (__bf16*)(ws + off); off += (size_t)256 * 512 * 2;

    k_cvt_u   <<<(ROWS * 256 + 255) / 256, 256, 0, stream>>>(x, ubf);
    k_cvt_win <<<(NPAD1 * 256 + 255) / 256, 256, 0, stream>>>(W_in, WinT);
    k_cvt_wout<<<(256 * 512 + 255) / 256, 256, 0, stream>>>(W_out, WoutT);

    k_inproj<<<dim3(19, 250), 128, 0, stream>>>(ubf, WinT, zxbcdt);
    k_conv<<<(ROWS * D_XBC + 255) / 256, 256, 0, stream>>>(zxbcdt, conv_w, conv_b, xbc);
    k_dt<<<(ROWS * NHEADS + 255) / 256, 256, 0, stream>>>(zxbcdt, dt_b, dts);
    k_scan<<<512, 256, 0, stream>>>(xbc, dts, A_log, Dp, yscan);
    k_gatenorm<<<ROWS, 256, 0, stream>>>(yscan, zxbcdt, nw, ynorm);
    k_outproj<<<dim3(4, 250), 128, 0, stream>>>(ynorm, WoutT, out);
}
